// ChannelAttention_13572096655430
// MI455X (gfx1250) — matrix operand order repaired and checked by reading, not compiled, not run on hardware
//
#include <hip/hip_runtime.h>
#include <cstdint>

typedef __bf16 bf16_t;
typedef __attribute__((ext_vector_type(16))) __bf16 v16bf;
typedef __attribute__((ext_vector_type(8)))  __bf16 v8bf;
typedef __attribute__((ext_vector_type(4)))  __bf16 v4bf;
typedef __attribute__((ext_vector_type(8)))  float  v8f;
typedef __attribute__((ext_vector_type(4)))  int    v4i;

#define AS1 __attribute__((address_space(1)))
#define AS3 __attribute__((address_space(3)))

#if defined(__HIP_DEVICE_COMPILE__) && defined(__has_builtin)
#if __has_builtin(__builtin_amdgcn_global_load_async_to_lds_b128)
#define HAVE_ASYNC_LDS 1
#endif
#endif
#ifndef HAVE_ASYNC_LDS
#define HAVE_ASYNC_LDS 0
#endif

__device__ __forceinline__ v8f wmma_bf16(v16bf a, v16bf b, v8f c) {
  return __builtin_amdgcn_wmma_f32_16x16x32_bf16(false, a, false, b, (short)0, c,
                                                 false, false);
}

// A fragment (16-bit 16x32): lane L -> row L%16, K = {s..s+7, s+16..s+23}, s=8*(L>=16)
__device__ __forceinline__ v16bf frag_a(const bf16_t* row_ptr, int k0, int lane) {
  int s = (lane >= 16) ? 8 : 0;
  v8bf lo = *(const v8bf*)(row_ptr + k0 + s);
  v8bf hi = *(const v8bf*)(row_ptr + k0 + s + 16);
  v16bf r;
#pragma unroll
  for (int i = 0; i < 8; ++i) { r[i] = lo[i]; r[i + 8] = hi[i]; }
  return r;
}

// B fragment (16-bit 32x16): lane L -> col L%16, K = {s..s+7, s+16..s+23}, s=8*(L>=16)
__device__ __forceinline__ v16bf frag_b(const bf16_t* col_ptr, int k0, int lane) {
  int t = (lane >= 16) ? 8 : 0;
  v8bf lo = *(const v8bf*)(col_ptr + k0 + t);
  v8bf hi = *(const v8bf*)(col_ptr + k0 + t + 16);
  v16bf r;
#pragma unroll
  for (int i = 0; i < 8; ++i) { r[i] = lo[i]; r[i + 8] = hi[i]; }
  return r;
}

// 16B stage: async copy global->LDS (ASYNCcnt path) with sync fallback.
__device__ __forceinline__ void stage_b128(const bf16_t* g, bf16_t* l) {
#if HAVE_ASYNC_LDS
  __builtin_amdgcn_global_load_async_to_lds_b128(
      (AS1 v4i*)(uintptr_t)g, (AS3 v4i*)(uint32_t)(uintptr_t)l, 0, 0);
#else
  *(v8bf*)l = *(const v8bf*)g;
#endif
}

__device__ __forceinline__ void stage_wait() {
#if HAVE_ASYNC_LDS
#if defined(__has_builtin) && __has_builtin(__builtin_amdgcn_s_wait_asynccnt)
  __builtin_amdgcn_s_wait_asynccnt(0);
#else
  asm volatile("s_wait_asynccnt 0" ::: "memory");
#endif
#endif
}

// ---------------------------------------------------------------- convert
__global__ __launch_bounds__(256) void cvt_f32_bf16(const float* __restrict__ src,
                                                    bf16_t* __restrict__ dst, int n4) {
  int i = blockIdx.x * blockDim.x + threadIdx.x;
  if (i < n4) {
    float4 v = ((const float4*)src)[i];
    v4bf p;
    p[0] = (bf16_t)v.x; p[1] = (bf16_t)v.y; p[2] = (bf16_t)v.z; p[3] = (bf16_t)v.w;
    ((v4bf*)dst)[i] = p;
  }
}

// ---------------------------------------------------------------- qkv GEMM
// C1[32768, 2304] = Xb[32768,768] * Wqkv^T. Block tile 128x128, 8 waves (4x2),
// wave tile 32x64 (2x4 WMMA). Double-buffered LDS, async staging.
// Epilogue scatters into q_t/k_t [bg,96,4096] (q scaled) and v_n [bg,4096,96].
__global__ __launch_bounds__(256) void gemm_qkv(const bf16_t* __restrict__ xb,
                                                const bf16_t* __restrict__ wb,
                                                bf16_t* __restrict__ qt,
                                                bf16_t* __restrict__ kt,
                                                bf16_t* __restrict__ vn) {
  __shared__ __align__(64) bf16_t lA[2][128 * 32];   // [m][k]
  __shared__ __align__(64) bf16_t lB[2][128 * 32];   // [n][k]
  const int tid = threadIdx.x, lane = tid & 31, wave = tid >> 5;
  const int wm = wave >> 1, wn = wave & 1;
  const int bm = blockIdx.x * 128, bn = blockIdx.y * 128;
  const int row = tid >> 1, col = (tid & 1) * 16;    // 16 elems/thread per tile
  v8f acc[2][4] = {};

  auto load_tile = [&](int k0, int buf) {
    const bf16_t* ga = xb + (size_t)(bm + row) * 768 + k0 + col;
    stage_b128(ga,     &lA[buf][row * 32 + col]);
    stage_b128(ga + 8, &lA[buf][row * 32 + col + 8]);
    const bf16_t* gb = wb + (size_t)(bn + row) * 768 + k0 + col;
    stage_b128(gb,     &lB[buf][row * 32 + col]);
    stage_b128(gb + 8, &lB[buf][row * 32 + col + 8]);
  };

  load_tile(0, 0);
  for (int k0 = 0, it = 0; k0 < 768; k0 += 32, ++it) {
    const int cur = it & 1;
    stage_wait();
    __syncthreads();
    if (k0 + 32 < 768) load_tile(k0 + 32, cur ^ 1);
    v16bf af[2], bfr[4];
#pragma unroll
    for (int t = 0; t < 2; ++t)
      af[t] = frag_a(&lA[cur][(wm * 32 + t * 16 + (lane & 15)) * 32], 0, lane);
#pragma unroll
    for (int u = 0; u < 4; ++u)
      bfr[u] = frag_b(&lB[cur][(wn * 64 + u * 16 + (lane & 15)) * 32], 0, lane);
#pragma unroll
    for (int tm = 0; tm < 2; ++tm)
#pragma unroll
      for (int tn = 0; tn < 4; ++tn) acc[tm][tn] = wmma_bf16(af[tm], bfr[tn], acc[tm][tn]);
    __syncthreads();
  }

  const int hi = (lane >> 4) * 8;
  const float qscale = 0.015625f;  // 4096^-0.5
#pragma unroll
  for (int tm = 0; tm < 2; ++tm) {
    int gm0 = bm + wm * 32 + tm * 16 + hi;          // rows gm0..gm0+7 share b
    int bb = gm0 >> 12, n0 = gm0 & 4095;
#pragma unroll
    for (int tn = 0; tn < 4; ++tn) {
      int gd = bn + wn * 64 + tn * 16 + (lane & 15);
      int three = gd / 768, rem = gd % 768;
      int g = rem / 96, c = rem % 96;
      v8f a = acc[tm][tn];
      if (three == 2) {
        bf16_t* dst = vn + ((size_t)(bb * 8 + g) * 4096 + n0) * 96 + c;
#pragma unroll
        for (int r = 0; r < 8; ++r) dst[(size_t)r * 96] = (bf16_t)a[r];
      } else {
        bf16_t* base = (three == 0) ? qt : kt;
        float s = (three == 0) ? qscale : 1.0f;
        v8bf pk;
#pragma unroll
        for (int r = 0; r < 8; ++r) pk[r] = (bf16_t)(a[r] * s);
        *(v8bf*)(base + ((size_t)((bb * 8 + g) * 96 + c)) * 4096 + n0) = pk;
      }
    }
  }
}

// ---------------------------------------------------------------- attention 1
// grid (64, 8): block (bg, kc) computes partial attn[96,96] over N-range
// [kc*512, kc*512+512). 9 waves (3x3), wave tile 32x32. Deterministic partials.
__global__ __launch_bounds__(288) void attn_partial(const bf16_t* __restrict__ qt,
                                                    const bf16_t* __restrict__ kt,
                                                    float* __restrict__ part) {
  const int bg = blockIdx.x, kc = blockIdx.y;
  const int tid = threadIdx.x, lane = tid & 31, wave = tid >> 5;
  const int wm = wave / 3, wn = wave % 3;
  const bf16_t* qrow = qt + (size_t)bg * 96 * 4096 +
                       (size_t)(wm * 32 + (lane & 15)) * 4096;
  const bf16_t* krow = kt + (size_t)bg * 96 * 4096 +
                       (size_t)(wn * 32 + (lane & 15)) * 4096;
  const int kbeg = kc * 512, kend = kbeg + 512;
  v8f acc[2][2] = {};
  for (int k0 = kbeg; k0 < kend; k0 += 32) {
    if (k0 + 128 < kend) {                     // stream-ahead hint
      __builtin_prefetch(qrow + k0 + 128, 0, 1);
      __builtin_prefetch(krow + k0 + 128, 0, 1);
    }
    v16bf af[2], bfr[2];
#pragma unroll
    for (int t = 0; t < 2; ++t) {
      af[t]  = frag_a(qrow + (size_t)t * 16 * 4096, k0, lane);
      bfr[t] = frag_b(krow + (size_t)t * 16 * 4096, k0, lane);
    }
#pragma unroll
    for (int tm = 0; tm < 2; ++tm)
#pragma unroll
      for (int tn = 0; tn < 2; ++tn) acc[tm][tn] = wmma_bf16(af[tm], bfr[tn], acc[tm][tn]);
  }
  const int hi = (lane >> 4) * 8;
  float* base = part + ((size_t)bg * 8 + kc) * 96 * 96;
#pragma unroll
  for (int tm = 0; tm < 2; ++tm)
#pragma unroll
    for (int tn = 0; tn < 2; ++tn) {
      int n = wn * 32 + tn * 16 + (lane & 15);
#pragma unroll
      for (int r = 0; r < 8; ++r)
        base[(size_t)(wm * 32 + tm * 16 + hi + r) * 96 + n] = acc[tm][tn][r];
    }
}

// ---------------------------------------------------------------- softmax
// grid 64 x 96 threads: thread = row c. Sum 8 partials, softmax over d, emit bf16.
__global__ __launch_bounds__(96) void softmax_rows(const float* __restrict__ part,
                                                   bf16_t* __restrict__ attnB) {
  const int bg = blockIdx.x, c = threadIdx.x;
  const float* p = part + (size_t)bg * 8 * 96 * 96 + (size_t)c * 96;
  float row[96];
#pragma unroll
  for (int d = 0; d < 96; ++d) {
    float s = 0.f;
#pragma unroll
    for (int ch = 0; ch < 8; ++ch) s += p[(size_t)ch * 96 * 96 + d];
    row[d] = s;
  }
  float mx = -3.4e38f;
#pragma unroll
  for (int d = 0; d < 96; ++d) mx = fmaxf(mx, row[d]);
  float sum = 0.f;
#pragma unroll
  for (int d = 0; d < 96; ++d) { row[d] = __expf(row[d] - mx); sum += row[d]; }
  float inv = 1.0f / sum;
  bf16_t* o = attnB + (size_t)bg * 96 * 96 + (size_t)c * 96;
#pragma unroll
  for (int d = 0; d < 96; ++d) o[d] = (bf16_t)(row[d] * inv);
}

// ---------------------------------------------------------------- attention 2
// grid (64, 32): block (bg, nc) computes out[96, 128cols] = attn(96x96) * v^T,
// attn staged in LDS. 8 waves, each owns 16 columns, 6 m-tiles, K=96.
__global__ __launch_bounds__(256) void attn_out(const bf16_t* __restrict__ attnB,
                                                const bf16_t* __restrict__ vn,
                                                bf16_t* __restrict__ aout) {
  __shared__ __align__(64) bf16_t sA[96 * 96];       // 18 KB
  const int bg = blockIdx.x, nc = blockIdx.y;
  const int b = bg >> 3, g = bg & 7;
  const int tid = threadIdx.x, lane = tid & 31, wave = tid >> 5;
  {
    const v8bf* src = (const v8bf*)(attnB + (size_t)bg * 96 * 96);
    v8bf* dst = (v8bf*)sA;
    for (int i = tid; i < (96 * 96) / 8; i += 256) dst[i] = src[i];
  }
  __syncthreads();
  const int ncol = nc * 128 + wave * 16 + (lane & 15);
  const bf16_t* vrow = vn + ((size_t)bg * 4096 + ncol) * 96;
  v8f oacc[6] = {};
#pragma unroll
  for (int ks = 0; ks < 3; ++ks) {
    const int k0 = ks * 32;
    v16bf bfr = frag_b(vrow, k0, lane);
#pragma unroll
    for (int mt = 0; mt < 6; ++mt) {
      v16bf af = frag_a(sA + (mt * 16 + (lane & 15)) * 96, k0, lane);
      oacc[mt] = wmma_bf16(af, bfr, oacc[mt]);
    }
  }
  const int hi = (lane >> 4) * 8;
  bf16_t* dst = aout + ((size_t)b * 4096 + ncol) * 768 + g * 96;
#pragma unroll
  for (int mt = 0; mt < 6; ++mt) {
    v8bf pk;
#pragma unroll
    for (int r = 0; r < 8; ++r) pk[r] = (bf16_t)oacc[mt][r];
    *(v8bf*)(dst + mt * 16 + hi) = pk;
  }
}

// ---------------------------------------------------------------- projection
// Y[32768,768] = Ab[32768,768] * Wproj^T + bias, fp32 out. Same tiling as qkv.
__global__ __launch_bounds__(256) void gemm_proj(const bf16_t* __restrict__ ab,
                                                 const bf16_t* __restrict__ wb,
                                                 const float* __restrict__ bias,
                                                 float* __restrict__ out) {
  __shared__ __align__(64) bf16_t lA[2][128 * 32];
  __shared__ __align__(64) bf16_t lB[2][128 * 32];
  const int tid = threadIdx.x, lane = tid & 31, wave = tid >> 5;
  const int wm = wave >> 1, wn = wave & 1;
  const int bm = blockIdx.x * 128, bn = blockIdx.y * 128;
  const int row = tid >> 1, col = (tid & 1) * 16;
  v8f acc[2][4] = {};

  auto load_tile = [&](int k0, int buf) {
    const bf16_t* ga = ab + (size_t)(bm + row) * 768 + k0 + col;
    stage_b128(ga,     &lA[buf][row * 32 + col]);
    stage_b128(ga + 8, &lA[buf][row * 32 + col + 8]);
    const bf16_t* gb = wb + (size_t)(bn + row) * 768 + k0 + col;
    stage_b128(gb,     &lB[buf][row * 32 + col]);
    stage_b128(gb + 8, &lB[buf][row * 32 + col + 8]);
  };

  load_tile(0, 0);
  for (int k0 = 0, it = 0; k0 < 768; k0 += 32, ++it) {
    const int cur = it & 1;
    stage_wait();
    __syncthreads();
    if (k0 + 32 < 768) load_tile(k0 + 32, cur ^ 1);
    v16bf af[2], bfr[4];
#pragma unroll
    for (int t = 0; t < 2; ++t)
      af[t] = frag_a(&lA[cur][(wm * 32 + t * 16 + (lane & 15)) * 32], 0, lane);
#pragma unroll
    for (int u = 0; u < 4; ++u)
      bfr[u] = frag_b(&lB[cur][(wn * 64 + u * 16 + (lane & 15)) * 32], 0, lane);
#pragma unroll
    for (int tm = 0; tm < 2; ++tm)
#pragma unroll
      for (int tn = 0; tn < 4; ++tn) acc[tm][tn] = wmma_bf16(af[tm], bfr[tn], acc[tm][tn]);
    __syncthreads();
  }

  const int hi = (lane >> 4) * 8;
#pragma unroll
  for (int tm = 0; tm < 2; ++tm) {
    int gm0 = bm + wm * 32 + tm * 16 + hi;
#pragma unroll
    for (int tn = 0; tn < 4; ++tn) {
      int gd = bn + wn * 64 + tn * 16 + (lane & 15);
      float bv = bias[gd];
      v8f a = acc[tm][tn];
#pragma unroll
      for (int r = 0; r < 8; ++r) out[(size_t)(gm0 + r) * 768 + gd] = a[r] + bv;
    }
  }
}

// ---------------------------------------------------------------- launch
extern "C" void kernel_launch(void* const* d_in, const int* in_sizes, int n_in,
                              void* d_out, int out_size, void* d_ws, size_t ws_size,
                              hipStream_t stream) {
  (void)in_sizes; (void)n_in; (void)out_size; (void)ws_size;
  const float* x      = (const float*)d_in[0];
  const float* w_qkv  = (const float*)d_in[1];
  const float* w_proj = (const float*)d_in[2];
  const float* b_proj = (const float*)d_in[3];
  float* out = (float*)d_out;

  char* ws = (char*)d_ws;
  size_t off = 0;
  auto alloc = [&](size_t bytes) -> void* {
    void* p = ws + off;
    off += (bytes + 255) & ~(size_t)255;
    return p;
  };
  bf16_t* xb   = (bf16_t*)alloc((size_t)32768 * 768 * 2);
  bf16_t* wqb  = (bf16_t*)alloc((size_t)2304 * 768 * 2);
  bf16_t* wpb  = (bf16_t*)alloc((size_t)768 * 768 * 2);
  bf16_t* qt   = (bf16_t*)alloc((size_t)64 * 96 * 4096 * 2);
  bf16_t* kt   = (bf16_t*)alloc((size_t)64 * 96 * 4096 * 2);
  bf16_t* vn   = (bf16_t*)alloc((size_t)64 * 4096 * 96 * 2);
  float*  part = (float*)alloc((size_t)64 * 8 * 96 * 96 * 4);
  bf16_t* atb  = (bf16_t*)alloc((size_t)64 * 96 * 96 * 2);
  bf16_t* ab   = (bf16_t*)alloc((size_t)32768 * 768 * 2);

  int n4;
  n4 = 32768 * 768 / 4;
  cvt_f32_bf16<<<(n4 + 255) / 256, 256, 0, stream>>>(x, xb, n4);
  n4 = 2304 * 768 / 4;
  cvt_f32_bf16<<<(n4 + 255) / 256, 256, 0, stream>>>(w_qkv, wqb, n4);
  n4 = 768 * 768 / 4;
  cvt_f32_bf16<<<(n4 + 255) / 256, 256, 0, stream>>>(w_proj, wpb, n4);

  gemm_qkv<<<dim3(256, 18), 256, 0, stream>>>(xb, wqb, qt, kt, vn);
  attn_partial<<<dim3(64, 8), 288, 0, stream>>>(qt, kt, part);
  softmax_rows<<<64, 96, 0, stream>>>(part, atb);
  attn_out<<<dim3(64, 32), 256, 0, stream>>>(atb, vn, ab);
  gemm_proj<<<dim3(256, 6), 256, 0, stream>>>(ab, wpb, b_proj, out);
}
